// DeformableTransformerDecoderLayer2_24429773980417
// MI455X (gfx1250) — matrix operand order repaired and checked by reading, not compiled, not run on hardware
//
#include <hip/hip_runtime.h>

typedef __attribute__((ext_vector_type(16))) _Float16 v16h;
typedef __attribute__((ext_vector_type(8)))  float    v8f;

#define E_   128
#define D_   256
#define N_   384

// ---------------------------------------------------------------------------
// Generic WMMA GEMM:  C[M,N] = act( alpha * (A[M,K] @ W[N,K]^T + bias) )
// one wave per 16x16 output tile; f16 inputs, f32 accumulate.
// ---------------------------------------------------------------------------
__global__ void gemm16(const float* __restrict__ A, int lda,
                       const float* __restrict__ W, int ldw,
                       const float* __restrict__ bias,
                       float* __restrict__ C, int ldc,
                       int K, float alpha, int relu)
{
    int lane = threadIdx.x;
    int ml   = lane & 15;
    int half = lane >> 4;
    int mg   = blockIdx.y * 16 + ml;   // row this lane feeds for A-frag
    int ng   = blockIdx.x * 16 + ml;   // col this lane feeds for B-frag
    const float* Arow = A + (size_t)mg * lda;
    const float* Wrow = W + (size_t)ng * ldw;

    v8f acc = {};
    for (int kk = 0; kk < K; kk += 32) {
        v16h a, b;
        const float* a1 = Arow + kk + 16 * half;       // K = kk+16h .. +8
        const float* a2 = Arow + kk + 16 * half + 8;   // K = kk+16h+8 .. +8
#pragma unroll
        for (int j = 0; j < 8; ++j) {
            a[j]     = (_Float16)a1[j];
            a[8 + j] = (_Float16)a2[j];
        }
        const float* b1 = Wrow + kk + 16 * half;       // 16 contiguous K per lane
#pragma unroll
        for (int j = 0; j < 16; ++j) b[j] = (_Float16)b1[j];
        acc = __builtin_amdgcn_wmma_f32_16x16x32_f16(false, a, false, b,
                                                     (short)0, acc, false, false);
    }

    float bn = bias ? bias[ng] : 0.f;
#pragma unroll
    for (int v = 0; v < 8; ++v) {
        int m = blockIdx.y * 16 + v + 8 * half;
        float val = (acc[v] + bn) * alpha;
        if (relu) val = fmaxf(val, 0.f);
        C[(size_t)m * ldc + ng] = val;
    }
}

// ---------------------------------------------------------------------------
// Self-attention per head: scores = Q@K^T (WMMA 16x16x32, DH=32 => 1 step),
// softmax in LDS, out = P@V (WMMA, K=128). Q pre-scaled by DH^-0.5.
// grid = (8 mtiles, 8 heads), block = 32 (one wave).
// ---------------------------------------------------------------------------
__global__ void attn_kernel(const float* __restrict__ Q,
                            const float* __restrict__ Km,
                            const float* __restrict__ Vm,
                            float* __restrict__ sa)
{
    __shared__ float sc[16][128];
    int lane  = threadIdx.x;
    int mtile = blockIdx.x;
    int h     = blockIdx.y;
    int ml    = lane & 15;
    int half  = lane >> 4;

    // A-fragment from Q (16 rows x 32 K)
    const float* qrow = Q + (size_t)(mtile * 16 + ml) * D_ + h * 32;
    v16h aq;
#pragma unroll
    for (int j = 0; j < 8; ++j) {
        aq[j]     = (_Float16)qrow[16 * half + j];
        aq[8 + j] = (_Float16)qrow[16 * half + 8 + j];
    }
#pragma unroll
    for (int nt = 0; nt < 8; ++nt) {
        const float* krow = Km + (size_t)(nt * 16 + ml) * D_ + h * 32 + 16 * half;
        v16h bk;
#pragma unroll
        for (int j = 0; j < 16; ++j) bk[j] = (_Float16)krow[j];
        v8f d = {};
        d = __builtin_amdgcn_wmma_f32_16x16x32_f16(false, aq, false, bk,
                                                   (short)0, d, false, false);
#pragma unroll
        for (int v = 0; v < 8; ++v) sc[v + 8 * half][nt * 16 + ml] = d[v];
    }
    __syncthreads();

    if (lane < 16) {
        float mx = -1e30f;
        for (int c = 0; c < 128; ++c) mx = fmaxf(mx, sc[lane][c]);
        float s = 0.f;
        for (int c = 0; c < 128; ++c) { float e = __expf(sc[lane][c] - mx); sc[lane][c] = e; s += e; }
        float inv = 1.0f / s;
        for (int c = 0; c < 128; ++c) sc[lane][c] *= inv;
    }
    __syncthreads();

    v8f o0 = {}, o1 = {};
    for (int kk = 0; kk < 4; ++kk) {
        v16h ap;
#pragma unroll
        for (int j = 0; j < 8; ++j) {
            ap[j]     = (_Float16)sc[ml][kk * 32 + 16 * half + j];
            ap[8 + j] = (_Float16)sc[ml][kk * 32 + 16 * half + 8 + j];
        }
        v16h bv0, bv1;
#pragma unroll
        for (int j = 0; j < 16; ++j) {
            size_t kr = (size_t)(kk * 32 + 16 * half + j) * D_ + h * 32;
            bv0[j] = (_Float16)Vm[kr + ml];
            bv1[j] = (_Float16)Vm[kr + 16 + ml];
        }
        o0 = __builtin_amdgcn_wmma_f32_16x16x32_f16(false, ap, false, bv0, (short)0, o0, false, false);
        o1 = __builtin_amdgcn_wmma_f32_16x16x32_f16(false, ap, false, bv1, (short)0, o1, false, false);
    }
#pragma unroll
    for (int v = 0; v < 8; ++v) {
        size_t row = (size_t)(mtile * 16 + v + 8 * half) * D_ + h * 32;
        sa[row + ml]      = o0[v];
        sa[row + 16 + ml] = o1[v];
    }
}

// ---------------------------------------------------------------------------
__global__ void add_kernel(const float* __restrict__ a, const float* __restrict__ b,
                           float* __restrict__ c)
{
    int i = blockIdx.x * blockDim.x + threadIdx.x;
    c[i] = a[i] + b[i];
}

// LayerNorm over rows of 256: out = LN(a + b) ; pool3: b rows averaged over 3.
__global__ void ln_kernel(const float* __restrict__ a, const float* __restrict__ b,
                          int pool3, const float* __restrict__ w,
                          const float* __restrict__ bb, float* __restrict__ out)
{
    __shared__ float red[256];
    int r = blockIdx.x, t = threadIdx.x;
    float v = a[(size_t)r * 256 + t];
    if (b) {
        if (pool3)
            v += (b[(size_t)(r * 3 + 0) * 256 + t] +
                  b[(size_t)(r * 3 + 1) * 256 + t] +
                  b[(size_t)(r * 3 + 2) * 256 + t]) * (1.0f / 3.0f);
        else
            v += b[(size_t)r * 256 + t];
    }
    red[t] = v; __syncthreads();
    for (int s = 128; s > 0; s >>= 1) { if (t < s) red[t] += red[t + s]; __syncthreads(); }
    float mean = red[0] * (1.0f / 256.0f);
    __syncthreads();
    float d = v - mean;
    red[t] = d * d; __syncthreads();
    for (int s = 128; s > 0; s >>= 1) { if (t < s) red[t] += red[t + s]; __syncthreads(); }
    float var = red[0] * (1.0f / 256.0f);
    out[(size_t)r * 256 + t] = d * rsqrtf(var + 1e-5f) * w[t] + bb[t];
}

// ---------------------------------------------------------------------------
// Per-point geometry: interpolation, tile window, segment clip, sinusoidal PE,
// query feature concat  ->  nqin[N,640]; also ref[N,2] and per-level (lx,ly).
// ---------------------------------------------------------------------------
__global__ void point_setup(const float* __restrict__ xln, const float* __restrict__ qpos,
                            const float* __restrict__ edge, float* __restrict__ nqin,
                            float* __restrict__ refb, int* __restrict__ lxly)
{
    int n = blockIdx.x * blockDim.x + threadIdx.x;
    if (n >= N_) return;
    int e = n / 3, j = n - e * 3;
    float ax = edge[e * 4 + 0], ay = edge[e * 4 + 1];
    float bx = edge[e * 4 + 2], by = edge[e * 4 + 3];
    float dx = bx - ax, dy = by - ay;
    float t = (float)j * 0.5f;
    float ptx = ax + t * dx, pty = ay + t * dy;
    int cxi = (int)floorf(ptx), cyi = (int)floorf(pty);
    int minx = cxi - 128; if (minx < 0) minx = 0; if (minx + 256 > 2048) minx = 2048 - 256;
    int miny = cyi - 128; if (miny < 0) miny = 0; if (miny + 256 > 2048) miny = 2048 - 256;
    float fminx = (float)minx, fminy = (float)miny;

    float tlx, thx, tly, thy;
    {
        float safe = (dx == 0.f) ? 1.f : dx;
        float t1 = (fminx - ax) / safe, t2 = (fminx + 256.f - ax) / safe;
        tlx = (dx == 0.f) ? 0.f : fminf(t1, t2);
        thx = (dx == 0.f) ? 1.f : fmaxf(t1, t2);
    }
    {
        float safe = (dy == 0.f) ? 1.f : dy;
        float t1 = (fminy - ay) / safe, t2 = (fminy + 256.f - ay) / safe;
        tly = (dy == 0.f) ? 0.f : fminf(t1, t2);
        thy = (dy == 0.f) ? 1.f : fmaxf(t1, t2);
    }
    float t0  = fmaxf(fmaxf(tlx, tly), 0.f);
    float t1v = fmaxf(fminf(fminf(thx, thy), 1.f), t0);
    float posx[3] = { ax + t0 * dx, ax + t1v * dx, (float)cxi };
    float posy[3] = { ay + t0 * dy, ay + t1v * dy, (float)cyi };

    refb[n * 2 + 0] = ((float)cxi - fminx) * (1.0f / 256.0f);
    refb[n * 2 + 1] = ((float)cyi - fminy) * (1.0f / 256.0f);
    const int Ws[4] = {256, 128, 64, 32};
    for (int l = 0; l < 4; ++l) {
        float ratio = (float)(2048 / Ws[l]);
        lxly[n * 8 + l * 2 + 0] = (int)rintf(fminx / ratio);
        lxly[n * 8 + l * 2 + 1] = (int)rintf(fminy / ratio);
    }

    float* dst = nqin + (size_t)n * 640;
    for (int c = 0; c < 256; ++c) dst[c] = xln[(size_t)e * 256 + c] + qpos[(size_t)e * 256 + c];
    for (int p = 0; p < 3; ++p)
        for (int d = 0; d < 128; ++d) {
            float v  = (d < 64) ? posy[p] : posx[p];
            int   dd = d & 63;
            int   m  = dd >> 1;
            float freq = __powf(10000.f, (float)m * (1.0f / 32.0f));
            float arg  = v / freq;
            dst[256 + p * 128 + d] = (dd & 1) ? __cosf(arg) : __sinf(arg);
        }
}

__global__ void aw_softmax(const float* __restrict__ attw, float* __restrict__ aw)
{
    int n = blockIdx.x, t = threadIdx.x;
    if (t < 8) {
        const float* s0 = attw + (size_t)n * 128 + t * 16;
        float mx = -1e30f;
        for (int i = 0; i < 16; ++i) mx = fmaxf(mx, s0[i]);
        float ex[16], s = 0.f;
        for (int i = 0; i < 16; ++i) { ex[i] = __expf(s0[i] - mx); s += ex[i]; }
        float inv = 1.0f / s;
        float* d = aw + (size_t)n * 128 + t * 16;
        for (int i = 0; i < 16; ++i) d[i] = ex[i] * inv;
    }
}

// ---------------------------------------------------------------------------
// Deformable sampling with the linearity trick:
//   acc_src[n,h,:] = sum_taps w * src[tap,:],  accW = sum w  (mask/bounds aware)
//   caIn[n,h*32+dh] = acc_src . val_w_row + accW * val_b
// grid = N, block = 256 (8 waves = 8 heads; lane owns 8 contiguous channels).
// ---------------------------------------------------------------------------
__global__ void sample_kernel(const float* __restrict__ src, const unsigned char* __restrict__ mask,
                              const float* __restrict__ vr, const float* __restrict__ refb,
                              const int* __restrict__ lxly, const float* __restrict__ off,
                              const float* __restrict__ aw, const float* __restrict__ valw,
                              const float* __restrict__ valb, float* __restrict__ caIn)
{
    __shared__ float accS[8][256];
    __shared__ float accWs[8];
    int n = blockIdx.x, tid = threadIdx.x;
    int h = tid >> 5, lane = tid & 31;
    const int Sl[4] = {32, 16, 8, 4};
    const int Wl[4] = {256, 128, 64, 32};
    const int St[4] = {0, 65536, 81920, 86016};

    float acc[8];
#pragma unroll
    for (int i = 0; i < 8; ++i) acc[i] = 0.f;
    float accW = 0.f;
    float rx = refb[n * 2], ry = refb[n * 2 + 1];

    for (int l = 0; l < 4; ++l) {
        int s = Sl[l]; float fs = (float)s;
        int wl = Wl[l]; int start = St[l];
        float lrx = rx * vr[l * 2 + 0];
        float lry = ry * vr[l * 2 + 1];
        int lx = lxly[n * 8 + l * 2 + 0];
        int ly = lxly[n * 8 + l * 2 + 1];
        for (int p = 0; p < 4; ++p) {
            int oidx = n * 256 + ((h * 4 + l) * 4 + p) * 2;
            float ox = off[oidx], oy = off[oidx + 1];
            float a_w = aw[n * 128 + h * 16 + l * 4 + p];
            float px = (lrx + ox / fs) * fs - 0.5f;
            float py = (lry + oy / fs) * fs - 0.5f;
            float fx0 = floorf(px), fy0 = floorf(py);
            int x0 = (int)fx0, y0 = (int)fy0;
            float fx = px - fx0, fy = py - fy0;
            float wts[4] = {(1.f - fx) * (1.f - fy), fx * (1.f - fy),
                            (1.f - fx) * fy,         fx * fy};
            int xs[4] = {x0, x0 + 1, x0, x0 + 1};
            int ys[4] = {y0, y0, y0 + 1, y0 + 1};
#pragma unroll
            for (int tp = 0; tp < 4; ++tp) {
                int xi = xs[tp], yi = ys[tp];
                if (xi >= 0 && xi < s && yi >= 0 && yi < s) {
                    long gidx = (long)start + (long)(ly + yi) * wl + (lx + xi);
                    if (!mask[gidx]) {
                        float wt = a_w * wts[tp];
                        const float* sp = src + gidx * 256 + lane * 8;
                        float4 v0 = *(const float4*)sp;
                        float4 v1 = *(const float4*)(sp + 4);
                        acc[0] += wt * v0.x; acc[1] += wt * v0.y;
                        acc[2] += wt * v0.z; acc[3] += wt * v0.w;
                        acc[4] += wt * v1.x; acc[5] += wt * v1.y;
                        acc[6] += wt * v1.z; acc[7] += wt * v1.w;
                        accW += wt;
                    }
                }
            }
        }
    }
#pragma unroll
    for (int i = 0; i < 8; ++i) accS[h][lane * 8 + i] = acc[i];
    if (lane == 0) accWs[h] = accW;
    __syncthreads();

    // head projection: output channel o = tid, head = tid/32
    const float* wrow = valw + (size_t)tid * 256;
    const float* as = accS[tid >> 5];
    float sum = 0.f;
    for (int c = 0; c < 256; ++c) sum += as[c] * wrow[c];
    caIn[(size_t)n * 256 + tid] = sum + accWs[tid >> 5] * valb[tid];
}

// ---------------------------------------------------------------------------
extern "C" void kernel_launch(void* const* d_in, const int* in_sizes, int n_in,
                              void* d_out, int out_size, void* d_ws, size_t ws_size,
                              hipStream_t stream)
{
    const float* tgt    = (const float*)d_in[0];
    const float* qpos   = (const float*)d_in[1];
    const float* edge   = (const float*)d_in[2];
    const float* srcf   = (const float*)d_in[3];
    const unsigned char* maskp = (const unsigned char*)d_in[4];
    const float* vrp    = (const float*)d_in[5];
    const float* inw    = (const float*)d_in[6];
    const float* inb    = (const float*)d_in[7];
    const float* outw   = (const float*)d_in[8];
    const float* outb   = (const float*)d_in[9];
    const float* n1w    = (const float*)d_in[10];
    const float* n1b    = (const float*)d_in[11];
    const float* n2w    = (const float*)d_in[12];
    const float* n2b    = (const float*)d_in[13];
    const float* n3w    = (const float*)d_in[14];
    const float* n3b    = (const float*)d_in[15];
    const float* lin0w  = (const float*)d_in[16];
    const float* lin0b  = (const float*)d_in[17];
    const float* lin1w  = (const float*)d_in[18];
    const float* lin1b  = (const float*)d_in[19];
    const float* lin2w  = (const float*)d_in[20];
    const float* lin2b  = (const float*)d_in[21];
    const float* offw   = (const float*)d_in[22];
    const float* offb   = (const float*)d_in[23];
    const float* attww  = (const float*)d_in[24];
    const float* attwb  = (const float*)d_in[25];
    const float* valw   = (const float*)d_in[26];
    const float* valb   = (const float*)d_in[27];
    const float* oprojw = (const float*)d_in[28];
    const float* oprojb = (const float*)d_in[29];

    float* ws   = (float*)d_ws;
    float* qk    = ws;            ws += 32768;
    float* Qb    = ws;            ws += 32768;
    float* Kb    = ws;            ws += 32768;
    float* Vb    = ws;            ws += 32768;
    float* sa    = ws;            ws += 32768;
    float* saP   = ws;            ws += 32768;
    float* x     = ws;            ws += 32768;
    float* nqin  = ws;            ws += 245760;
    float* nq    = ws;            ws += 98304;
    float* offB  = ws;            ws += 98304;
    float* attwB = ws;            ws += 49152;
    float* awB   = ws;            ws += 49152;
    float* refb  = ws;            ws += 768;
    float* caIn  = ws;            ws += 98304;
    float* caOut = ws;            ws += 98304;
    float* x2    = ws;            ws += 32768;
    float* h1    = ws;            ws += 131072;
    float* ffn   = ws;            ws += 32768;
    int*   lxly  = (int*)ws;

    const float qscale = 0.17677669529663689f; // 32^-0.5

    // 1. qk = tgt + query_pos
    add_kernel<<<128, 256, 0, stream>>>(tgt, qpos, qk);
    // 2. Q,K,V projections (q scaled)
    gemm16<<<dim3(16, 8), 32, 0, stream>>>(qk, 256, inw,            256, inb,       Qb, 256, 256, qscale, 0);
    gemm16<<<dim3(16, 8), 32, 0, stream>>>(qk, 256, inw + 256 * 256,256, inb + 256, Kb, 256, 256, 1.f,    0);
    gemm16<<<dim3(16, 8), 32, 0, stream>>>(tgt,256, inw + 512 * 256,256, inb + 512, Vb, 256, 256, 1.f,    0);
    // 3. self-attention
    attn_kernel<<<dim3(8, 8), 32, 0, stream>>>(Qb, Kb, Vb, sa);
    // 4. out_proj, residual + LN2 -> x
    gemm16<<<dim3(16, 8), 32, 0, stream>>>(sa, 256, outw, 256, outb, saP, 256, 256, 1.f, 0);
    ln_kernel<<<128, 256, 0, stream>>>(saP, tgt, 0, n2w, n2b, x);
    // 5. per-point geometry + PE + qfeat concat
    point_setup<<<3, 128, 0, stream>>>(x, qpos, edge, nqin, refb, lxly);
    // 6. lin0 / offsets / attention weights
    gemm16<<<dim3(16, 24), 32, 0, stream>>>(nqin, 640, lin0w, 640, lin0b, nq,    256, 640, 1.f, 0);
    gemm16<<<dim3(16, 24), 32, 0, stream>>>(nq,   256, offw,  256, offb,  offB,  256, 256, 1.f, 0);
    gemm16<<<dim3(8, 24),  32, 0, stream>>>(nq,   256, attww, 256, attwb, attwB, 128, 256, 1.f, 0);
    aw_softmax<<<384, 32, 0, stream>>>(attwB, awB);
    // 7. deformable sampling (linearity trick) + head projection
    sample_kernel<<<384, 256, 0, stream>>>(srcf, maskp, vrp, refb, lxly, offB, awB, valw, valb, caIn);
    // 8. oproj, pool over PTS, residual + LN1 -> x2
    gemm16<<<dim3(16, 24), 32, 0, stream>>>(caIn, 256, oprojw, 256, oprojb, caOut, 256, 256, 1.f, 0);
    ln_kernel<<<128, 256, 0, stream>>>(x, caOut, 1, n1w, n1b, x2);
    // 9. FFN + final LN -> d_out
    gemm16<<<dim3(64, 8), 32, 0, stream>>>(x2, 256,  lin1w, 256,  lin1b, h1,  1024, 256,  1.f, 1);
    gemm16<<<dim3(16, 8), 32, 0, stream>>>(h1, 1024, lin2w, 1024, lin2b, ffn, 256,  1024, 1.f, 0);
    ln_kernel<<<128, 256, 0, stream>>>(x2, ffn, 0, n3w, n3b, (float*)d_out);
}
